// DiSAN_46119358824847
// MI455X (gfx1250) — matrix operand order repaired and checked by reading, not compiled, not run on hardware
//
#include <hip/hip_runtime.h>
#include <hip/hip_bf16.h>

typedef __attribute__((ext_vector_type(16))) __bf16 v16bf;
typedef __attribute__((ext_vector_type(8)))  __bf16 v8bf;
typedef __attribute__((ext_vector_type(8)))  float  v8f;

#define S_LEN 500
#define S_PAD 512
#define D_DIM 1024
#define B_DIM 64
#define M_ROWS (B_DIM * S_LEN)   // 32000
#define VT_STRIDE ((size_t)B_DIM * S_PAD)   // per-d row length of V^T

// ---------------------------------------------------------------------------
// Weight convert + transpose: W[K][N] f32  ->  Wt[N][K] bf16
// ---------------------------------------------------------------------------
__global__ void wconv_kernel(const float* __restrict__ W, __bf16* __restrict__ Wt,
                             int K, int N) {
  size_t idx = (size_t)blockIdx.x * 256 + threadIdx.x;
  if (idx >= (size_t)K * N) return;
  int k = (int)(idx / N);
  int n = (int)(idx % N);
  Wt[(size_t)n * K + k] = (__bf16)W[idx];
}

// ---------------------------------------------------------------------------
// H = x + pos_emb ; fp32 and bf16 copies
// ---------------------------------------------------------------------------
__global__ void addpos_kernel(const float* __restrict__ x, const float* __restrict__ pos,
                              float* __restrict__ Hf, __bf16* __restrict__ Hb) {
  size_t i4 = ((size_t)blockIdx.x * 256 + threadIdx.x) * 4;
  if (i4 >= (size_t)M_ROWS * D_DIM) return;
  size_t row = i4 >> 10;            // / 1024
  int d = (int)(i4 & 1023);
  int srow = (int)(row % S_LEN);
  float4 xv = *(const float4*)(x + i4);
  float4 pv = *(const float4*)(pos + (size_t)srow * D_DIM + d);
  float4 h;
  h.x = xv.x + pv.x; h.y = xv.y + pv.y; h.z = xv.z + pv.z; h.w = xv.w + pv.w;
  *(float4*)(Hf + i4) = h;
  Hb[i4 + 0] = (__bf16)h.x;
  Hb[i4 + 1] = (__bf16)h.y;
  Hb[i4 + 2] = (__bf16)h.z;
  Hb[i4 + 3] = (__bf16)h.w;
}

// ---------------------------------------------------------------------------
// Generic WMMA GEMM:  out = epi( A[M x K](bf16) * Wt[N][K](bf16) + bias )
// A is split at column `ksplit` between A0 and A1 (for concat([H, H_m])).
// EPI: 0 = store bf16 (val = (acc+bias)*alpha)
//      1 = store f32
//      2 = store bf16 sigmoid(val)
//      3 = gate: g=sigmoid(val); c = g*H + (1-g)*Hm ; store f32+bf16 at col ocol
//      4 = store bf16 TRANSPOSED into V^T [col][batch][seq-padded]
// Block: 256 threads = 8 waves; block tile 64x128; wave tile 16x64.
// ---------------------------------------------------------------------------
template<int EPI>
__global__ __launch_bounds__(256) void gemm_wmma_kernel(
    const __bf16* __restrict__ A0, const __bf16* __restrict__ A1, int ksplit,
    int lda0, int lda1,
    const __bf16* __restrict__ Wt, const float* __restrict__ bias,
    int N, int K, float alpha,
    __bf16* __restrict__ outB, float* __restrict__ outF, int ocol,
    const float* __restrict__ Hf, const __bf16* __restrict__ Hm) {
  const int t = threadIdx.x;
  const int wave = t >> 5, lane = t & 31;
  const int nlane = lane & 15, hi = lane >> 4;
  const int wm = wave & 3, wn = wave >> 2;
  const int row0 = blockIdx.y * 64 + wm * 16;
  const int col0 = blockIdx.x * 128 + wn * 64;

  v8f zero = {};
  v8f acc[4];
#pragma unroll
  for (int i = 0; i < 4; ++i) acc[i] = zero;

  for (int k0 = 0; k0 < K; k0 += 32) {
    const __bf16* Abase; int kk, lda;
    if (k0 < ksplit) { Abase = A0; kk = k0;          lda = lda0; }
    else             { Abase = A1; kk = k0 - ksplit; lda = lda1; }
    // A fragment: lane m=nlane holds K = kk+hi*16..+15
    const __bf16* pa = Abase + (size_t)(row0 + nlane) * lda + kk + hi * 16;
    v16bf a;
    {
      v8bf x0 = *(const v8bf*)pa;
      v8bf x1 = *(const v8bf*)(pa + 8);
#pragma unroll
      for (int j = 0; j < 8; ++j) { a[j] = x0[j]; a[j + 8] = x1[j]; }
    }
    __builtin_prefetch((const void*)(pa + 64), 0, 1);
#pragma unroll
    for (int i = 0; i < 4; ++i) {
      // B fragment from transposed weights: lane n holds 16 contiguous K values
      const __bf16* pw = Wt + (size_t)(col0 + 16 * i + nlane) * K + k0 + hi * 16;
      v16bf bfr = *(const v16bf*)pw;
      acc[i] = __builtin_amdgcn_wmma_f32_16x16x32_bf16(
          false, a, false, bfr, (short)0, acc[i], false, false);
    }
  }

  // Epilogue: D element (reg r, lane) -> row = row0 + r + 8*hi, col = col0+16i+nlane
#pragma unroll
  for (int i = 0; i < 4; ++i) {
    int col = col0 + 16 * i + nlane;
    float bcol = bias[col];
#pragma unroll
    for (int r = 0; r < 8; ++r) {
      int row = row0 + r + 8 * hi;
      float val = (acc[i][r] + bcol) * alpha;
      if (EPI == 0) {
        outB[(size_t)row * N + col] = (__bf16)val;
      } else if (EPI == 1) {
        outF[(size_t)row * N + col] = val;
      } else if (EPI == 2) {
        float g = 1.f / (1.f + __expf(-val));
        outB[(size_t)row * N + col] = (__bf16)g;
      } else if (EPI == 3) {
        float g  = 1.f / (1.f + __expf(-val));
        float h  = Hf[(size_t)row * D_DIM + col];
        float hm = (float)Hm[(size_t)row * D_DIM + col];
        float c  = g * h + (1.f - g) * hm;
        outF[(size_t)row * 2048 + ocol + col] = c;
        outB[(size_t)row * 2048 + ocol + col] = (__bf16)c;
      } else { // EPI == 4: V^T store, key dim padded to S_PAD per batch
        int bb = row / S_LEN;
        int s  = row - bb * S_LEN;
        outB[(size_t)col * VT_STRIDE + (size_t)bb * S_PAD + s] = (__bf16)val;
      }
    }
  }
}

// ---------------------------------------------------------------------------
// Flash attention with causal (dir=0) / anti-causal (dir=1) mask.
// Grid: (S/16 q-tiles, B, 2). Block: 256 threads = 8 waves.
// Q scaled by 1/sqrt(D) at projection time. V consumed pre-transposed from
// global (so PV B-fragments are contiguous 32B loads) -> only ~45KB LDS,
// several workgroups per WGP. Q tile staged with async global->LDS copies.
// ---------------------------------------------------------------------------
__global__ __launch_bounds__(256) void attn_kernel(
    const __bf16* __restrict__ Qf, const __bf16* __restrict__ Kf, const __bf16* __restrict__ Vtf,
    const __bf16* __restrict__ Qb, const __bf16* __restrict__ Kb, const __bf16* __restrict__ Vtb,
    __bf16* __restrict__ HmF, __bf16* __restrict__ HmB) {
  __shared__ __attribute__((aligned(32))) __bf16 qs[16 * D_DIM];     // Q tile, row-major
  __shared__ __attribute__((aligned(32))) float  sc[16 * 128];       // raw scores
  __shared__ __attribute__((aligned(32))) __bf16 pm[16 * 128];       // probabilities (bf16)
  __shared__ float rowmax[16], rowsum[16], rowscale[16];
  __shared__ float pmax[16][16], psum[16][16];

  const int dir = blockIdx.z;
  const int b   = blockIdx.y;
  const int q0  = blockIdx.x * 16;
  const __bf16* Q  = dir ? Qb  : Qf;
  const __bf16* Km = dir ? Kb  : Kf;
  const __bf16* Vt = dir ? Vtb : Vtf;
  __bf16* Hm = dir ? HmB : HmF;
  const size_t rowbase = (size_t)b * S_LEN;

  const int t = threadIdx.x;
  const int wave = t >> 5, lane = t & 31;
  const int nlane = lane & 15, hi = lane >> 4;

  // ---- stage Q tile ----
  if (q0 + 16 <= S_LEN) {
    // fast path: async DMA global -> LDS (no VGPR round-trip, ASYNCcnt tracked)
    const __bf16* Qrow = Q + (rowbase + q0) * (size_t)D_DIM;
    unsigned long long qbase = (unsigned long long)(uintptr_t)Qrow;
    for (int i = t; i < 16 * (D_DIM / 8); i += 256) {
      unsigned elem = (unsigned)i * 8u;                    // element offset in tile
      unsigned ldsAddr = (unsigned)(uintptr_t)&qs[elem];   // LDS byte offset (low bits)
      unsigned gOff = elem * 2u;                           // global byte offset
      asm volatile("global_load_async_to_lds_b128 %0, %1, %2"
                   :: "v"(ldsAddr), "v"(gOff), "s"(qbase)
                   : "memory");
    }
  } else {
    // boundary tile: bounds-checked vector loads with zero pad
    for (int i = t; i < 16 * (D_DIM / 8); i += 256) {
      int row  = i >> 7;
      int dpos = (i & 127) * 8;
      v8bf v = {};
      if (q0 + row < S_LEN)
        v = *(const v8bf*)(Q + (rowbase + q0 + row) * (size_t)D_DIM + dpos);
      *(v8bf*)&qs[(size_t)row * D_DIM + dpos] = v;
    }
  }
  asm volatile("s_wait_asynccnt 0x0" ::: "memory");
  if (t < 16) { rowmax[t] = -1e30f; rowsum[t] = 0.f; }

  v8f zero = {};
  v8f ao[8];
#pragma unroll
  for (int cc = 0; cc < 8; ++cc) ao[cc] = zero;

  int kbFirst, kbLast, kbStep;
  if (dir == 0) { kbFirst = 0;                 kbLast = (q0 + 15) / 128; kbStep = 1;  }
  else          { kbFirst = (S_LEN - 1) / 128; kbLast = q0 / 128;        kbStep = -1; }

  __syncthreads();

  for (int kbi = kbFirst;; kbi += kbStep) {
    const int kb = kbi * 128;

    // ---- scores: each wave handles 16 keys, full-D reduction (32 wmma) ----
    v8f acc = zero;
    const int keyBase = kb + wave * 16;
    const bool keyOk = (keyBase + nlane) < S_LEN;
    const __bf16* krow = Km + (rowbase + keyBase + nlane) * (size_t)D_DIM;
    for (int kc = 0; kc < 32; ++kc) {
      v16bf a;
      {
        const __bf16* p = &qs[(size_t)nlane * D_DIM + kc * 32 + hi * 16];
#pragma unroll
        for (int j = 0; j < 8; ++j) { a[j] = p[j]; a[j + 8] = p[j + 8]; }
      }
      v16bf bfr = {};
      if (keyOk) bfr = *(const v16bf*)(krow + kc * 32 + hi * 16);
      acc = __builtin_amdgcn_wmma_f32_16x16x32_bf16(
          false, a, false, bfr, (short)0, acc, false, false);
    }
#pragma unroll
    for (int r = 0; r < 8; ++r) {
      int lrow = r + 8 * hi;
      int q    = q0 + lrow;
      int key  = keyBase + nlane;
      bool keep = (q < S_LEN) && (key < S_LEN) && (dir ? (key >= q) : (key <= q));
      sc[lrow * 128 + wave * 16 + nlane] = keep ? acc[r] : -1e30f;
    }
    __syncthreads();

    // ---- online softmax: block max ----
    {
      int row = t & 15, seg = t >> 4;
      float m = -1e30f;
#pragma unroll
      for (int j = 0; j < 8; ++j) m = fmaxf(m, sc[row * 128 + seg * 8 + j]);
      pmax[row][seg] = m;
    }
    __syncthreads();
    if (t < 16) {
      float bm = -1e30f;
      for (int j = 0; j < 16; ++j) bm = fmaxf(bm, pmax[t][j]);
      float mo = rowmax[t];
      float mn = fmaxf(mo, bm);
      rowscale[t] = __expf(mo - mn);
      rowmax[t]   = mn;
    }
    __syncthreads();

    // ---- p = exp(s - m), partial sums, rescale output accumulators ----
    {
      int row = t & 15, seg = t >> 4;
      float mn = rowmax[row];
      float s = 0.f;
#pragma unroll
      for (int j = 0; j < 8; ++j) {
        float p = __expf(sc[row * 128 + seg * 8 + j] - mn);
        pm[row * 128 + seg * 8 + j] = (__bf16)p;
        s += p;
      }
      psum[row][seg] = s;
    }
#pragma unroll
    for (int r = 0; r < 8; ++r) {
      float scl = rowscale[r + 8 * hi];
#pragma unroll
      for (int cc = 0; cc < 8; ++cc) ao[cc][r] *= scl;
    }
    __syncthreads();
    if (t < 16) {
      float s = 0.f;
      for (int j = 0; j < 16; ++j) s += psum[t][j];
      rowsum[t] = rowsum[t] * rowscale[t] + s;
    }

    // ---- PV: wave owns 128 output columns; B-frags straight from global V^T ----
    const __bf16* vbb = Vt + (size_t)b * S_PAD + kb;
#pragma unroll
    for (int cc = 0; cc < 8; ++cc) {
#pragma unroll
      for (int kc = 0; kc < 4; ++kc) {
        v16bf a;
        {
          const __bf16* p = &pm[(size_t)nlane * 128 + kc * 32 + hi * 16];
#pragma unroll
          for (int j = 0; j < 8; ++j) { a[j] = p[j]; a[j + 8] = p[j + 8]; }
        }
        const int dcol = wave * 128 + cc * 16 + nlane;
        v16bf bfr = *(const v16bf*)(vbb + (size_t)dcol * VT_STRIDE + kc * 32 + hi * 16);
        ao[cc] = __builtin_amdgcn_wmma_f32_16x16x32_bf16(
            false, a, false, bfr, (short)0, ao[cc], false, false);
      }
    }
    __syncthreads();
    if (kbi == kbLast) break;
  }

  // ---- epilogue: out = acc / rowsum ----
#pragma unroll
  for (int r = 0; r < 8; ++r) {
    int lrow = r + 8 * hi;
    int q = q0 + lrow;
    if (q < S_LEN) {
      float inv = 1.f / rowsum[lrow];
#pragma unroll
      for (int cc = 0; cc < 8; ++cc) {
        int dcol = wave * 128 + cc * 16 + nlane;
        Hm[(rowbase + q) * (size_t)D_DIM + dcol] = (__bf16)(ao[cc][r] * inv);
      }
    }
  }
}

// ---------------------------------------------------------------------------
// utt_vec: softmax over seq dim of Sc2[B,S,2048], weighted sum of C_i
// ---------------------------------------------------------------------------
__global__ void uttvec_kernel(const float* __restrict__ Sc2, const float* __restrict__ Ci,
                              float* __restrict__ out) {
  int idx = blockIdx.x * 256 + threadIdx.x;
  if (idx >= B_DIM * 2048) return;
  int b = idx >> 11, c = idx & 2047;
  const float* sp = Sc2 + ((size_t)b * S_LEN) * 2048 + c;
  const float* cp = Ci  + ((size_t)b * S_LEN) * 2048 + c;
  float m = -1e30f;
  for (int s = 0; s < S_LEN; ++s) m = fmaxf(m, sp[(size_t)s * 2048]);
  float sum = 0.f;
  for (int s = 0; s < S_LEN; ++s) sum += __expf(sp[(size_t)s * 2048] - m);
  float acc = 0.f;
  for (int s = 0; s < S_LEN; ++s)
    acc += __expf(sp[(size_t)s * 2048] - m) * cp[(size_t)s * 2048];
  out[idx] = acc / sum;
}

// ---------------------------------------------------------------------------
extern "C" void kernel_launch(void* const* d_in, const int* in_sizes, int n_in,
                              void* d_out, int out_size, void* d_ws, size_t ws_size,
                              hipStream_t stream) {
  (void)in_sizes; (void)n_in; (void)out_size; (void)ws_size;

  const float* x_in  = (const float*)d_in[0];
  const float* pos   = (const float*)d_in[1];
  // Weights/biases: Wq_fw=2,bq_fw=3, Wk_fw=4,bk=5, Wv_fw=6,bv=7,
  //                 Wq_bw=8..13, Wg_fw=14,bg_fw=15, Wg_bw=16,bg_bw=17,
  //                 W2=18,b2=19, W1=20,b1=21

  const size_t HE  = (size_t)M_ROWS * D_DIM;       // 32,768,000
  const size_t CE  = (size_t)M_ROWS * 2048;
  const size_t VTE = (size_t)D_DIM * VT_STRIDE;    // 1024 * 64 * 512

  char* w = (char*)d_ws;
  size_t off = 0;
  auto take = [&](size_t bytes) -> char* {
    char* p = w + off;
    off += (bytes + 255) & ~(size_t)255;
    return p;
  };
  float*  Hf  = (float*) take(HE * 4);
  __bf16* Hb  = (__bf16*)take(HE * 2);
  __bf16* Qf  = (__bf16*)take(HE * 2);
  __bf16* Kf  = (__bf16*)take(HE * 2);
  __bf16* Qb  = (__bf16*)take(HE * 2);
  __bf16* Kb  = (__bf16*)take(HE * 2);
  __bf16* Vtf = (__bf16*)take(VTE * 2);
  __bf16* Vtb = (__bf16*)take(VTE * 2);
  __bf16* HmF = (__bf16*)take(HE * 2);
  __bf16* HmB = (__bf16*)take(HE * 2);
  __bf16* CiB = (__bf16*)take(CE * 2);
  __bf16* WT[10];
  const int wK[10] = {1024,1024,1024,1024,1024,1024, 2048,2048, 2048,2048};
  const int wN[10] = {1024,1024,1024,1024,1024,1024, 1024,1024, 2048,2048};
  const int wIdx[10] = {2,4,6,8,10,12, 14,16, 18,20};
  for (int i = 0; i < 10; ++i) WT[i] = (__bf16*)take((size_t)wK[i] * wN[i] * 2);

  // buffer reuse after attention:
  //   activ (bf16, 131MB) overlays Qf+Kf; Sc2 (f32, 262MB) overlays Qb..Vtb
  __bf16* activ = Qf;
  float*  Sc2   = (float*)Qb;

  float* utt = (float*)d_out;
  float* Ci  = (float*)d_out + (size_t)B_DIM * 2048;

  // 1) weight convert/transpose
  for (int i = 0; i < 10; ++i) {
    size_t n = (size_t)wK[i] * wN[i];
    wconv_kernel<<<dim3((unsigned)((n + 255) / 256)), 256, 0, stream>>>(
        (const float*)d_in[wIdx[i]], WT[i], wK[i], wN[i]);
  }

  // 2) H = x + pos
  {
    size_t n4 = HE / 4;
    addpos_kernel<<<dim3((unsigned)((n4 + 255) / 256)), 256, 0, stream>>>(x_in, pos, Hf, Hb);
  }

  // 3) projections (Q scaled by 1/sqrt(D)=1/32; V stored transposed)
  dim3 pgrid(1024 / 128, M_ROWS / 64);
  gemm_wmma_kernel<0><<<pgrid, 256, 0, stream>>>(Hb, Hb, 1024, 1024, 1024, WT[0],
      (const float*)d_in[3], 1024, 1024, 0.03125f, Qf, nullptr, 0, nullptr, nullptr);
  gemm_wmma_kernel<0><<<pgrid, 256, 0, stream>>>(Hb, Hb, 1024, 1024, 1024, WT[1],
      (const float*)d_in[5], 1024, 1024, 1.0f, Kf, nullptr, 0, nullptr, nullptr);
  gemm_wmma_kernel<4><<<pgrid, 256, 0, stream>>>(Hb, Hb, 1024, 1024, 1024, WT[2],
      (const float*)d_in[7], 1024, 1024, 1.0f, Vtf, nullptr, 0, nullptr, nullptr);
  gemm_wmma_kernel<0><<<pgrid, 256, 0, stream>>>(Hb, Hb, 1024, 1024, 1024, WT[3],
      (const float*)d_in[9], 1024, 1024, 0.03125f, Qb, nullptr, 0, nullptr, nullptr);
  gemm_wmma_kernel<0><<<pgrid, 256, 0, stream>>>(Hb, Hb, 1024, 1024, 1024, WT[4],
      (const float*)d_in[11], 1024, 1024, 1.0f, Kb, nullptr, 0, nullptr, nullptr);
  gemm_wmma_kernel<4><<<pgrid, 256, 0, stream>>>(Hb, Hb, 1024, 1024, 1024, WT[5],
      (const float*)d_in[13], 1024, 1024, 1.0f, Vtb, nullptr, 0, nullptr, nullptr);

  // 4) attention (fw causal, bw anti-causal)
  attn_kernel<<<dim3(S_PAD / 16, B_DIM, 2), 256, 0, stream>>>(
      Qf, Kf, Vtf, Qb, Kb, Vtb, HmF, HmB);

  // 5) gates -> C_i (f32 to d_out, bf16 to ws)
  gemm_wmma_kernel<3><<<pgrid, 256, 0, stream>>>(
      Hb, HmF, 1024, 1024, 1024, WT[6], (const float*)d_in[15],
      1024, 2048, 1.0f, CiB, Ci, 0, Hf, HmF);
  gemm_wmma_kernel<3><<<pgrid, 256, 0, stream>>>(
      Hb, HmB, 1024, 1024, 1024, WT[7], (const float*)d_in[17],
      1024, 2048, 1.0f, CiB, Ci, 1024, Hf, HmB);

  // 6) activ = sigmoid(C_i @ W2 + b2)   [bf16]
  gemm_wmma_kernel<2><<<dim3(2048 / 128, M_ROWS / 64), 256, 0, stream>>>(
      CiB, CiB, 2048, 2048, 2048, WT[8], (const float*)d_in[19],
      2048, 2048, 1.0f, activ, nullptr, 0, nullptr, nullptr);

  // 7) scores = activ @ W1 + b1   [f32]
  gemm_wmma_kernel<1><<<dim3(2048 / 128, M_ROWS / 64), 256, 0, stream>>>(
      activ, activ, 2048, 2048, 2048, WT[9], (const float*)d_in[21],
      2048, 2048, 1.0f, nullptr, Sc2, 0, nullptr, nullptr);

  // 8) utt_vec
  uttvec_kernel<<<dim3((B_DIM * 2048) / 256), 256, 0, stream>>>(Sc2, Ci, utt);
}
